// STGCN_1623497638414
// MI455X (gfx1250) — matrix operand order repaired and checked by reading, not compiled, not run on hardware
//
#include <hip/hip_runtime.h>

// ---------------------------------------------------------------------------
// Problem constants (from reference)
// ---------------------------------------------------------------------------
#define BB   256          // batch
#define NN   512          // nodes
#define TT   12           // time steps in X
#define FF   12           // features == LSTM sequence length
#define HID  1536         // 3*N
#define IN_  6144         // N*F
#define G4   6144         // 4*HID
#define MBIG (FF * BB)    // 3072 rows of the hoisted input GEMM

typedef __attribute__((ext_vector_type(16))) __bf16 v16bf;
typedef __attribute__((ext_vector_type(8)))  float  v8f;

// Async global->LDS staging (CDNA5). Diagnostic-confirmed parameter types:
//   (int4 addrspace(1)*, int4 addrspace(3)*, imm offset, imm cpol)
#if defined(__AMDGCN__) && __has_builtin(__builtin_amdgcn_global_load_async_to_lds_b128)
#define USE_ASYNC_LDS 1
typedef __attribute__((__vector_size__(16))) int i32x4;
typedef __attribute__((address_space(1))) i32x4 AsG128;   // global-AS int4
typedef __attribute__((address_space(3))) i32x4 AsL128;   // LDS-AS int4
#define ASYNC_CP16(gsrc, ldst) \
  __builtin_amdgcn_global_load_async_to_lds_b128((AsG128*)(gsrc), (AsL128*)(ldst), 0, 0)
#if __has_builtin(__builtin_amdgcn_s_wait_asynccnt)
#define WAIT_ASYNC(n) __builtin_amdgcn_s_wait_asynccnt(n)
#else
#define WAIT_ASYNC(n) asm volatile("s_wait_asynccnt %0" ::"i"(n) : "memory")
#endif
#else
#define USE_ASYNC_LDS 0
#endif

// ---------------------------------------------------------------------------
// Prep kernels
// ---------------------------------------------------------------------------
__global__ __launch_bounds__(256) void cvt_f32_bf16(const float* __restrict__ src,
                                                    __bf16* __restrict__ dst, int n) {
  int i = blockIdx.x * blockDim.x + threadIdx.x;
  if (i < n) dst[i] = (__bf16)src[i];
}

__global__ __launch_bounds__(256) void bias_combine(const float* __restrict__ a,
                                                    const float* __restrict__ b,
                                                    float* __restrict__ o, int n) {
  int i = blockIdx.x * blockDim.x + threadIdx.x;
  if (i < n) o[i] = a[i] + b[i];
}

// X (B,N,T,F) -> Xseq[(t*B + b)][n*T + tt] in bf16 (rows of the big GEMM)
__global__ __launch_bounds__(256) void pack_x(const float* __restrict__ X,
                                              __bf16* __restrict__ dst) {
  int idx = blockIdx.x * blockDim.x + threadIdx.x;    // < MBIG * IN_
  int j = idx % IN_;
  int r = idx / IN_;
  int t = r >> 8;            // / BB (BB == 256)
  int b = r & 255;
  int n  = j / TT;
  int tt = j % TT;
  size_t src = (((size_t)b * NN + n) * TT + tt) * FF + t;
  dst[idx] = (__bf16)X[src];
}

__global__ __launch_bounds__(256) void zero_state(float* __restrict__ c,
                                                  __bf16* __restrict__ h, int n) {
  int i = blockIdx.x * blockDim.x + threadIdx.x;
  if (i < n) { c[i] = 0.0f; h[i] = (__bf16)0.0f; }
}

// ---------------------------------------------------------------------------
// bf16 WMMA GEMM:  C[M,N] = A[M,K] * W[N,K]^T  (+ addend + bias)
//   A row-major (lda), W row-major with rows = output columns (ldb == K stride)
//   128x128 block tile, 8 waves, each wave 32x64 = 2x4 wmma 16x16x32 tiles.
//   Async path: K staged 64-deep, double-buffered via global_load_async_to_lds.
// ---------------------------------------------------------------------------
#define TILE_M 128
#define TILE_N 128
#define AK     64          // async-path K depth per stage
#define LDK    72          // padded LDS row length (144 B, 16 B aligned, bank-skewed)
#define FK     32          // fallback-path K depth per stage

__global__ __launch_bounds__(256)
void gemm_bf16_wmma(const __bf16* __restrict__ A, int lda,
                    const __bf16* __restrict__ Bw, int ldb,
                    const float* __restrict__ addend,
                    const float* __restrict__ bias,
                    float* __restrict__ C, int ldc, int K) {
  const int tid  = threadIdx.x;
  const int lane = tid & 31;
  const int wv   = tid >> 5;          // 0..7
  const int wm   = (wv & 3) * 32;     // wave row offset inside tile
  const int wn   = (wv >> 2) * 64;    // wave col offset inside tile
  const int ln   = lane & 15;         // matrix row/col within 16
  const int lh   = lane >> 4;         // K-half selector per ISA layout

  const int bm = blockIdx.y * TILE_M;
  const int bn = blockIdx.x * TILE_N;

  v8f acc[2][4];
  #pragma unroll
  for (int i = 0; i < 2; i++)
    #pragma unroll
    for (int j = 0; j < 4; j++)
      #pragma unroll
      for (int e = 0; e < 8; e++) acc[i][j][e] = 0.0f;

  union Frag { v16bf v; uint4 q[2]; };

#if USE_ASYNC_LDS
  // ---- double-buffered async-DMA pipeline (ASYNCcnt-tracked), K=64/stage ----
  __shared__ __bf16 As[2][TILE_M][LDK];
  __shared__ __bf16 Bs[2][TILE_N][LDK];

  // staging: 4 threads/row, 32 B (16 bf16) each, rows {srow, srow+64}
  const int srow = tid >> 2;          // 0..63
  const int scol = (tid & 3) * 16;    // 0,16,32,48

  const __bf16* gA = A  + (size_t)(bm + srow) * lda + scol;
  const __bf16* gB = Bw + (size_t)(bn + srow) * ldb + scol;
  const size_t aRow64 = (size_t)64 * lda;
  const size_t bRow64 = (size_t)64 * ldb;

  // 8 async 16B copies per (A,B) K-tile into buffer `buf`
  auto issue = [&](int buf, const __bf16* ga, const __bf16* gb) {
    ASYNC_CP16(ga,              &As[buf][srow][scol]);
    ASYNC_CP16(ga + 8,          &As[buf][srow][scol + 8]);
    ASYNC_CP16(ga + aRow64,     &As[buf][srow + 64][scol]);
    ASYNC_CP16(ga + aRow64 + 8, &As[buf][srow + 64][scol + 8]);
    ASYNC_CP16(gb,              &Bs[buf][srow][scol]);
    ASYNC_CP16(gb + 8,          &Bs[buf][srow][scol + 8]);
    ASYNC_CP16(gb + bRow64,     &Bs[buf][srow + 64][scol]);
    ASYNC_CP16(gb + bRow64 + 8, &Bs[buf][srow + 64][scol + 8]);
  };

  // 16 WMMAs over the K=64 stage in buffer `cur`
  auto compute = [&](int cur) {
    #pragma unroll
    for (int ks = 0; ks < AK; ks += 32) {
      Frag af[2], bfr[4];
      #pragma unroll
      for (int i = 0; i < 2; i++) {
        const __bf16* p = &As[cur][wm + i * 16 + ln][ks];
        af[i].q[0] = *(const uint4*)(p + 8 * lh);
        af[i].q[1] = *(const uint4*)(p + 16 + 8 * lh);
      }
      #pragma unroll
      for (int j = 0; j < 4; j++) {
        const __bf16* p = &Bs[cur][wn + j * 16 + ln][ks];
        bfr[j].q[0] = *(const uint4*)(p + 8 * lh);
        bfr[j].q[1] = *(const uint4*)(p + 16 + 8 * lh);
      }
      #pragma unroll
      for (int i = 0; i < 2; i++)
        #pragma unroll
        for (int j = 0; j < 4; j++)
          acc[i][j] = __builtin_amdgcn_wmma_f32_16x16x32_bf16(
              false, af[i].v, false, bfr[j].v, (short)0, acc[i][j], false, false);
    }
  };

  const int nk = K / AK;              // >= 2 for all our shapes
  issue(0, gA, gB);
  gA += AK; gB += AK;

  // steady state: branch-free body (last iteration peeled)
  for (int kt = 0; kt < nk - 1; ++kt) {
    issue((kt & 1) ^ 1, gA, gB);      // DMA next tile into other buffer
    gA += AK; gB += AK;
    WAIT_ASYNC(8);                    // 8 in flight for next => current done
    __syncthreads();
    compute(kt & 1);
    __syncthreads();                  // all reads done before buffer reuse
  }
  WAIT_ASYNC(0);
  __syncthreads();
  compute((nk - 1) & 1);
#else
  // ---- fallback: register-staged LDS path (known-good) ----
  __shared__ __bf16 As[TILE_M][FK];
  __shared__ __bf16 Bs[TILE_N][FK];
  const int srow = tid >> 1;
  const int scol = (tid & 1) * 16;
  const __bf16* gA = A  + (size_t)(bm + srow) * lda + scol;
  const __bf16* gB = Bw + (size_t)(bn + srow) * ldb + scol;

  for (int k0 = 0; k0 < K; k0 += FK) {
    uint4 a0 = *(const uint4*)(gA);
    uint4 a1 = *(const uint4*)(gA + 8);
    uint4 b0 = *(const uint4*)(gB);
    uint4 b1 = *(const uint4*)(gB + 8);
    __builtin_prefetch(gA + FK, 0, 1);   // -> global_prefetch_b8
    __builtin_prefetch(gB + FK, 0, 1);
    __syncthreads();
    *(uint4*)&As[srow][scol]     = a0;
    *(uint4*)&As[srow][scol + 8] = a1;
    *(uint4*)&Bs[srow][scol]     = b0;
    *(uint4*)&Bs[srow][scol + 8] = b1;
    __syncthreads();
    gA += FK; gB += FK;

    Frag af[2], bfr[4];
    #pragma unroll
    for (int i = 0; i < 2; i++) {
      const __bf16* p = &As[wm + i * 16 + ln][0];
      af[i].q[0] = *(const uint4*)(p + 8 * lh);
      af[i].q[1] = *(const uint4*)(p + 16 + 8 * lh);
    }
    #pragma unroll
    for (int j = 0; j < 4; j++) {
      const __bf16* p = &Bs[wn + j * 16 + ln][0];
      bfr[j].q[0] = *(const uint4*)(p + 8 * lh);
      bfr[j].q[1] = *(const uint4*)(p + 16 + 8 * lh);
    }
    #pragma unroll
    for (int i = 0; i < 2; i++)
      #pragma unroll
      for (int j = 0; j < 4; j++)
        acc[i][j] = __builtin_amdgcn_wmma_f32_16x16x32_bf16(
            false, af[i].v, false, bfr[j].v, (short)0, acc[i][j], false, false);
  }
#endif

  // Epilogue. f32 C/D layout: lane n=ln; VGPR r -> row r + 8*lh.
  #pragma unroll
  for (int i = 0; i < 2; i++) {
    #pragma unroll
    for (int j = 0; j < 4; j++) {
      const int row0 = bm + wm + i * 16;
      const int col  = bn + wn + j * 16 + ln;
      #pragma unroll
      for (int r = 0; r < 8; r++) {
        const int row = row0 + r + 8 * lh;
        size_t off = (size_t)row * ldc + col;
        float v = acc[i][j][r];
        if (addend) v += addend[off];
        if (bias)   v += bias[col];
        C[off] = v;
      }
    }
  }
}

// ---------------------------------------------------------------------------
// LSTM cell elementwise (PyTorch gate order i,f,g,o; extra tanh on h)
// ---------------------------------------------------------------------------
__global__ __launch_bounds__(256)
void lstm_cell(const float* __restrict__ gates, float* __restrict__ c,
               float* __restrict__ h_out, __bf16* __restrict__ h_bf) {
  int idx = blockIdx.x * blockDim.x + threadIdx.x;   // < BB*HID
  int b = idx / HID, j = idx % HID;
  const float* g = gates + (size_t)b * G4;
  float ig = g[j], fg = g[HID + j], gg = g[2 * HID + j], og = g[3 * HID + j];
  float si = 1.0f / (1.0f + __expf(-ig));
  float sf = 1.0f / (1.0f + __expf(-fg));
  float so = 1.0f / (1.0f + __expf(-og));
  float cn = sf * c[idx] + si * tanhf(gg);
  float h  = tanhf(so * tanhf(cn));
  c[idx] = cn;
  h_out[idx] = h;
  h_bf[idx] = (__bf16)h;
}

// ---------------------------------------------------------------------------
// Final: out[b,n,to] = sum_k relu(hs[k%12][b][3n + k/12]) * lin_W[to][k] + lin_b[to]
// ---------------------------------------------------------------------------
__global__ __launch_bounds__(256)
void final_linear(const float* __restrict__ hs, const float* __restrict__ W,
                  const float* __restrict__ bias, float* __restrict__ out) {
  int idx = blockIdx.x * blockDim.x + threadIdx.x;   // < BB*NN*12
  int to = idx % 12;
  int bn = idx / 12;
  int n = bn % NN, b = bn / NN;
  float s = bias[to];
  #pragma unroll
  for (int k = 0; k < 36; k++) {
    int f  = k % 12;
    int hc = 3 * n + k / 12;
    float v = hs[((size_t)f * BB + b) * HID + hc];
    s += fmaxf(v, 0.0f) * W[to * 36 + k];
  }
  out[idx] = s;
}

// ---------------------------------------------------------------------------
// Launch
// ---------------------------------------------------------------------------
extern "C" void kernel_launch(void* const* d_in, const int* in_sizes, int n_in,
                              void* d_out, int out_size, void* d_ws, size_t ws_size,
                              hipStream_t stream) {
  // inputs: 0 A_hat (unused), 1 X, 2 W_ih, 3 W_hh, 4 b_ih, 5 b_hh, 6 lin_W, 7 lin_b
  const float* X     = (const float*)d_in[1];
  const float* W_ih  = (const float*)d_in[2];
  const float* W_hh  = (const float*)d_in[3];
  const float* b_ih  = (const float*)d_in[4];
  const float* b_hh  = (const float*)d_in[5];
  const float* lin_W = (const float*)d_in[6];
  const float* lin_b = (const float*)d_in[7];
  float* out = (float*)d_out;

  // workspace layout
  char* w = (char*)d_ws;
  size_t off = 0;
  auto carve = [&](size_t bytes) { void* p = w + off; off += (bytes + 255) & ~size_t(255); return p; };
  __bf16* Wih_bf = (__bf16*)carve((size_t)G4 * IN_ * 2);       // 75.5 MB
  __bf16* Whh_bf = (__bf16*)carve((size_t)G4 * HID * 2);       // 18.9 MB
  __bf16* Xseq   = (__bf16*)carve((size_t)MBIG * IN_ * 2);     // 37.7 MB
  float*  biasC  = (float*) carve((size_t)G4 * 4);
  float*  Gin    = (float*) carve((size_t)MBIG * G4 * 4);      // 75.5 MB
  float*  gates  = (float*) carve((size_t)BB * G4 * 4);        // 6.3 MB
  float*  cbuf   = (float*) carve((size_t)BB * HID * 4);
  __bf16* hbf    = (__bf16*)carve((size_t)BB * HID * 2);
  float*  hs     = (float*) carve((size_t)FF * BB * HID * 4);  // 18.9 MB

  // --- prep ---
  {
    int n = G4 * IN_;
    cvt_f32_bf16<<<(n + 255) / 256, 256, 0, stream>>>(W_ih, Wih_bf, n);
  }
  {
    int n = G4 * HID;
    cvt_f32_bf16<<<(n + 255) / 256, 256, 0, stream>>>(W_hh, Whh_bf, n);
  }
  bias_combine<<<(G4 + 255) / 256, 256, 0, stream>>>(b_ih, b_hh, biasC, G4);
  {
    int n = MBIG * IN_;
    pack_x<<<(n + 255) / 256, 256, 0, stream>>>(X, Xseq);
  }
  zero_state<<<(BB * HID + 255) / 256, 256, 0, stream>>>(cbuf, hbf, BB * HID);

  // --- hoisted input GEMM: Gin = Xseq @ W_ih^T   (3072 x 6144, K = 6144) ---
  {
    dim3 grid(G4 / TILE_N, MBIG / TILE_M);
    gemm_bf16_wmma<<<grid, 256, 0, stream>>>(Xseq, IN_, Wih_bf, IN_,
                                             nullptr, nullptr, Gin, G4, IN_);
  }

  // --- recurrent loop ---
  for (int t = 0; t < FF; t++) {
    dim3 grid(G4 / TILE_N, BB / TILE_M);
    gemm_bf16_wmma<<<grid, 256, 0, stream>>>(hbf, HID, Whh_bf, HID,
                                             Gin + (size_t)t * BB * G4, biasC,
                                             gates, G4, HID);
    lstm_cell<<<(BB * HID + 255) / 256, 256, 0, stream>>>(
        gates, cbuf, hs + (size_t)t * BB * HID, hbf);
  }

  // --- final linear ---
  {
    int n = BB * NN * 12;
    final_linear<<<(n + 255) / 256, 256, 0, stream>>>(hs, lin_W, lin_b, out);
  }
  (void)in_sizes; (void)n_in; (void)out_size; (void)ws_size;
}
